// Attention_49950469652818
// MI455X (gfx1250) — matrix operand order repaired and checked by reading, not compiled, not run on hardware
//
#include <hip/hip_runtime.h>
#include <hip/hip_bf16.h>
#include <math.h>

// Problem constants (match reference)
#define NBATCH 8
#define SEQ    1024
#define DMODEL 768
#define NHEAD  12
#define DHEAD  64
#define NTOK   (NBATCH * SEQ)   // 8192 token rows

typedef __attribute__((ext_vector_type(16))) __bf16 v16bf;
typedef __attribute__((ext_vector_type(8)))  __bf16 v8bf;
typedef __attribute__((ext_vector_type(8)))  float  v8f;

// ---------------------------------------------------------------------------
// bf16 <-> f32 helpers (round-to-nearest-even)
// ---------------------------------------------------------------------------
static __device__ __forceinline__ __bf16 f2bf(float f) {
  unsigned u = __float_as_uint(f);
  u += 0x7FFFu + ((u >> 16) & 1u);
  unsigned short hs = (unsigned short)(u >> 16);
  __bf16 r;
  __builtin_memcpy(&r, &hs, sizeof(r));
  return r;
}

// Pack two f32 into one dword of bf16s (lo -> [15:0], hi -> [31:16]), RNE.
static __device__ __forceinline__ unsigned packbf2(float lo, float hi) {
  unsigned ul = __float_as_uint(lo);
  ul += 0x7FFFu + ((ul >> 16) & 1u);
  unsigned uh = __float_as_uint(hi);
  uh += 0x7FFFu + ((uh >> 16) & 1u);
  return (ul >> 16) | (uh & 0xFFFF0000u);
}

static __device__ __forceinline__ v16bf combine8(v8bf lo, v8bf hi) {
  v16bf f;
#pragma unroll
  for (int i = 0; i < 8; ++i) { f[i] = lo[i]; f[8 + i] = hi[i]; }
  return f;
}

// ---------------------------------------------------------------------------
// WMMA fragment loaders per CDNA5 ISA 7.12.2 (wave32 layouts), vectorized:
// each lane's fragment data is contiguous 16-byte runs -> b128 loads.
// ---------------------------------------------------------------------------

// A matrix 16x32 bf16 from row-major [m][k], stride ld (16B-multiple).
static __device__ __forceinline__ v16bf load_frag_a(const __bf16* base, int ld, int lane) {
  const int m    = lane & 15;
  const int half = lane >> 4;
  const __bf16* p = base + m * ld + (half << 3);
  v8bf lo = *(const v8bf*)(p);        // K = 8h + 0..7
  v8bf hi = *(const v8bf*)(p + 16);   // K = 16 + 8h + 0..7
  return combine8(lo, hi);
}

// B matrix 32x16 bf16 from an [n][k] buffer (B[k][n] = base[n*ld + k]).
static __device__ __forceinline__ v16bf load_frag_b_nk(const __bf16* base, int ld, int lane) {
  const int n    = lane & 15;
  const int kOff = (lane >> 4) << 3;
  const __bf16* p = base + n * ld + kOff;
  v8bf lo = *(const v8bf*)(p);
  v8bf hi = *(const v8bf*)(p + 16);
  return combine8(lo, hi);
}

static __device__ __forceinline__ v8f wmma_bf16(v16bf a, v16bf b, v8f c) {
  return __builtin_amdgcn_wmma_f32_16x16x32_bf16(false, a, false, b, (short)0, c, false, false);
}

// CDNA5 async LDS<->global 128-bit ops (ASYNCcnt). LDS byte address = low 32
// bits of the generic address (aperture rule).
static __device__ __forceinline__ void async_copy_b128(const void* gptr, void* ldsptr) {
  unsigned lds = (unsigned)(size_t)ldsptr;
  asm volatile("global_load_async_to_lds_b128 %0, %1, off"
               :: "v"(lds), "v"(gptr) : "memory");
}
static __device__ __forceinline__ void async_store_b128(void* gptr, const void* ldsptr) {
  unsigned lds = (unsigned)(size_t)ldsptr;
  asm volatile("global_store_async_from_lds_b128 %0, %1, off"
               :: "v"(gptr), "v"(lds) : "memory");
}
static __device__ __forceinline__ void wait_dscnt0() {
  asm volatile("s_wait_dscnt 0x0" ::: "memory");
}
static __device__ __forceinline__ void wait_asynccnt0() {
  asm volatile("s_wait_asynccnt 0x0" ::: "memory");
}

// Vectorized f32 -> bf16 tile staging: one float4 load -> one uint2 LDS store.
static __device__ __forceinline__ void stage4(const float* __restrict__ g, __bf16* s) {
  float4 f = *(const float4*)g;
  uint2 p;
  p.x = packbf2(f.x, f.y);
  p.y = packbf2(f.z, f.w);
  *(uint2*)s = p;
}

// ---------------------------------------------------------------------------
// Kernel 1: QKV projections.  Out[n, j] = sum_m X[n,m] * W[j,m] + b[j]
// j = h*64+dh. Q scaled by 1/sqrt(DH). V written TRANSPOSED:
//   Vt[(b*DMODEL + j) * SEQ + s]  so the P*V B-fragments are contiguous.
// Q/K epilogue: LDS transpose + async b128 stores from LDS.
// ---------------------------------------------------------------------------
__global__ __launch_bounds__(256)
void qkv_proj_kernel(const float* __restrict__ x,
                     const float* __restrict__ WQ, const float* __restrict__ bQ,
                     const float* __restrict__ WK, const float* __restrict__ bK,
                     const float* __restrict__ WV, const float* __restrict__ bV,
                     __bf16* __restrict__ Qw, __bf16* __restrict__ Kw,
                     __bf16* __restrict__ Vt) {
  __shared__ __align__(16) __bf16 sA[128][40];
  __shared__ __align__(16) __bf16 sB[64][40];
  __shared__ __align__(16) __bf16 eb[8][16 * 72];   // per-wave 16x64 epilogue tile
  const int t    = threadIdx.x;
  const int lane = t & 31;
  const int wave = t >> 5;
  const int row0 = blockIdx.x * 128;
  const int col0 = blockIdx.y * 64;
  const int proj = blockIdx.z;

  const float* W    = (proj == 0) ? WQ : ((proj == 1) ? WK : WV);
  const float* bias = (proj == 0) ? bQ : ((proj == 1) ? bK : bV);
  __bf16*      out  = (proj == 0) ? Qw : Kw;
  const float scale = (proj == 0) ? 0.125f : 1.0f;  // 1/sqrt(64) folded into Q

  v8f acc[4];
#pragma unroll
  for (int i = 0; i < 4; ++i)
#pragma unroll
    for (int v = 0; v < 8; ++v) acc[i][v] = 0.0f;

  for (int kk = 0; kk < DMODEL; kk += 32) {
    // Stage X tile 128x32 (1024 float4 chunks, 4 per thread)
#pragma unroll
    for (int i = 0; i < 4; ++i) {
      int chunk = i * 256 + t;
      int r = chunk >> 3, c = (chunk & 7) << 2;
      stage4(&x[(size_t)(row0 + r) * DMODEL + kk + c], &sA[r][c]);
    }
    // Stage W tile 64x32 (512 chunks, 2 per thread)
#pragma unroll
    for (int i = 0; i < 2; ++i) {
      int chunk = i * 256 + t;
      int r = chunk >> 3, c = (chunk & 7) << 2;
      stage4(&W[(size_t)(col0 + r) * DMODEL + kk + c], &sB[r][c]);
    }
    if (kk + 32 < DMODEL) {  // prefetch next X tile -> global_prefetch_b8
      __builtin_prefetch(&x[(size_t)(row0 + (t >> 1)) * DMODEL + kk + 32 + (t & 1) * 16], 0, 1);
    }
    __syncthreads();

    v16bf a = load_frag_a(&sA[wave * 16][0], 40, lane);
#pragma unroll
    for (int nb = 0; nb < 4; ++nb) {
      v16bf b = load_frag_b_nk(&sB[nb * 16][0], 40, lane);
      acc[nb] = wmma_bf16(a, b, acc[nb]);
    }
    __syncthreads();
  }

  const int n = lane & 15, half = lane >> 4;
  const int rbase = row0 + wave * 16 + half * 8;   // multiple of 8
  if (proj == 2) {
    // V transposed: lane's 8 rows are contiguous in s -> one b128 store per nb
    int bb = rbase >> 10, s0 = rbase & (SEQ - 1);
#pragma unroll
    for (int nb = 0; nb < 4; ++nb) {
      int col  = col0 + nb * 16 + n;
      float bv = bias[col];
      v8bf pk;
#pragma unroll
      for (int v = 0; v < 8; ++v) pk[v] = f2bf(acc[nb][v] + bv);
      *(v8bf*)(Vt + ((size_t)bb * DMODEL + col) * SEQ + s0) = pk;
    }
  } else {
    // Q/K: C-layout -> LDS tile, then coalesced async b128 stores from LDS
#pragma unroll
    for (int nb = 0; nb < 4; ++nb) {
      int col  = col0 + nb * 16 + n;
      float bv = bias[col];
#pragma unroll
      for (int v = 0; v < 8; ++v) {
        eb[wave][(half * 8 + v) * 72 + nb * 16 + n] = f2bf((acc[nb][v] + bv) * scale);
      }
    }
    wait_dscnt0();   // async engine LDS reads are not ordered with DS writes
    int r = lane & 15, c0 = half << 5;   // each lane streams half a row (32 elems)
    __bf16* dst = out + (size_t)(row0 + wave * 16 + r) * DMODEL + col0 + c0;
    const __bf16* src = &eb[wave][r * 72 + c0];
#pragma unroll
    for (int i = 0; i < 4; ++i) async_store_b128(dst + 8 * i, src + 8 * i);
    wait_asynccnt0();
  }
}

// ---------------------------------------------------------------------------
// Kernel 2: flash attention (causal, online softmax).
// Block 128 = 4 independent waves; each wave owns 16 query rows of one (b,h).
// Per 32-key block: 4 WMMA for S = Q K^T, LDS relayout of P, 4 WMMA for P V.
// Z epilogue: LDS transpose + async b128 stores from LDS.
// ---------------------------------------------------------------------------
__global__ __launch_bounds__(128)
void flash_attn_kernel(const __bf16* __restrict__ Qw, const __bf16* __restrict__ Kw,
                       const __bf16* __restrict__ Vt, __bf16* __restrict__ Zw) {
  __shared__ __align__(16) __bf16 pbuf[4][16 * 72];  // P relayout + Z epilogue
  const int t = threadIdx.x, lane = t & 31, wave = t >> 5;
  const int b = blockIdx.z, h = blockIdx.y;
  const int q0   = blockIdx.x * 64 + wave * 16;
  const int n    = lane & 15;
  const int half = lane >> 4;
  const size_t tok0 = (size_t)b * SEQ;
  const int hoff = h * DHEAD;

  // Q tile 16x64 => two 16x32 A fragments (Q already holds the 1/sqrt(DH))
  const __bf16* qptr = Qw + (tok0 + q0) * DMODEL + hoff;
  v16bf aQ0 = load_frag_a(qptr, DMODEL, lane);
  v16bf aQ1 = load_frag_a(qptr + 32, DMODEL, lane);

  v8f accZ[4];
  float mrow[8], lrow[8];
#pragma unroll
  for (int v = 0; v < 8; ++v) {
    mrow[v] = -1e30f;
    lrow[v] = 0.0f;
#pragma unroll
    for (int nb = 0; nb < 4; ++nb) accZ[nb][v] = 0.0f;
  }

  const __bf16* vtb = Vt + ((size_t)b * DMODEL + hoff) * SEQ;  // [dh][s] for this (b,h)

  for (int j0 = 0; j0 <= q0 + 15; j0 += 32) {
    // ---- scores: 16 rows x 32 keys as two 16x16 C tiles ----
    v8f sc0, sc1;
#pragma unroll
    for (int v = 0; v < 8; ++v) { sc0[v] = 0.0f; sc1[v] = 0.0f; }

    const __bf16* kbase = Kw + (tok0 + j0) * DMODEL + hoff;
    {
      v16bf bk0 = load_frag_b_nk(kbase, DMODEL, lane);
      v16bf bk1 = load_frag_b_nk(kbase + 32, DMODEL, lane);
      sc0 = wmma_bf16(aQ0, bk0, sc0);
      sc0 = wmma_bf16(aQ1, bk1, sc0);
    }
    {
      const __bf16* kb = kbase + (size_t)16 * DMODEL;
      v16bf bk0 = load_frag_b_nk(kb, DMODEL, lane);
      v16bf bk1 = load_frag_b_nk(kb + 32, DMODEL, lane);
      sc1 = wmma_bf16(aQ0, bk0, sc1);
      sc1 = wmma_bf16(aQ1, bk1, sc1);
    }

    // ---- causal mask + online softmax (row = half*8+v, col = n (+16)) ----
#pragma unroll
    for (int v = 0; v < 8; ++v) {
      int row  = q0 + half * 8 + v;
      float s0 = (j0 + n      > row) ? -1e30f : sc0[v];
      float s1 = (j0 + 16 + n > row) ? -1e30f : sc1[v];

      float mx = fmaxf(s0, s1);
#pragma unroll
      for (int off = 8; off >= 1; off >>= 1) mx = fmaxf(mx, __shfl_xor(mx, off, 32));
      float mnew = fmaxf(mrow[v], mx);
      float resc = __expf(mrow[v] - mnew);
      float p0 = __expf(s0 - mnew);
      float p1 = __expf(s1 - mnew);
      float ps = p0 + p1;
#pragma unroll
      for (int off = 8; off >= 1; off >>= 1) ps += __shfl_xor(ps, off, 32);
      lrow[v] = lrow[v] * resc + ps;
      mrow[v] = mnew;
#pragma unroll
      for (int nb = 0; nb < 4; ++nb) accZ[nb][v] *= resc;

      // stash P (C-tile layout) into LDS for relayout into an A fragment
      pbuf[wave][(half * 8 + v) * 72 + n]      = f2bf(p0);
      pbuf[wave][(half * 8 + v) * 72 + 16 + n] = f2bf(p1);
    }
    wait_dscnt0();

    v16bf pA = load_frag_a(&pbuf[wave][0], 72, lane);  // P as 16x32 A fragment

    // ---- accZ += P (16x32) @ V (32x64); Vt gives [n][k] fragments ----
#pragma unroll
    for (int nb = 0; nb < 4; ++nb) {
      v16bf bv = load_frag_b_nk(vtb + (size_t)(nb * 16) * SEQ + j0, SEQ, lane);
      accZ[nb] = wmma_bf16(pA, bv, accZ[nb]);
    }
  }

  // ---- finalize: divide by row sums; LDS transpose; async b128 stores ----
#pragma unroll
  for (int v = 0; v < 8; ++v) {
    float inv = 1.0f / lrow[v];
#pragma unroll
    for (int nb = 0; nb < 4; ++nb) {
      pbuf[wave][(half * 8 + v) * 72 + nb * 16 + n] = f2bf(accZ[nb][v] * inv);
    }
  }
  wait_dscnt0();
  {
    int r = lane & 15, c0 = half << 5;
    __bf16* dst = Zw + (tok0 + q0 + r) * DMODEL + hoff + c0;
    const __bf16* src = &pbuf[wave][r * 72 + c0];
#pragma unroll
    for (int i = 0; i < 4; ++i) async_store_b128(dst + 8 * i, src + 8 * i);
  }
  wait_asynccnt0();
}

// ---------------------------------------------------------------------------
// Kernel 3: output projection. out[n, m] = sum_j Z[n, j] * W_O[j>>6, m, j&63] + b_O[m]
// A-tile staging: GLOBAL_LOAD_ASYNC_TO_LDS_B128. Epilogue: LDS transpose +
// GLOBAL_STORE_ASYNC_FROM_LDS_B128 (f32).
// ---------------------------------------------------------------------------
__global__ __launch_bounds__(256)
void out_proj_kernel(const __bf16* __restrict__ Zw, const float* __restrict__ WO,
                     const float* __restrict__ bO, float* __restrict__ out) {
  __shared__ __align__(16) __bf16 sA[128][40];
  __shared__ __align__(16) __bf16 sB[64][40];
  __shared__ __align__(16) float  ef[8][16 * 68];   // per-wave 16x64 f32 tile
  const int t = threadIdx.x, lane = t & 31, wave = t >> 5;
  const int row0 = blockIdx.x * 128;
  const int col0 = blockIdx.y * 64;

  v8f acc[4];
#pragma unroll
  for (int i = 0; i < 4; ++i)
#pragma unroll
    for (int v = 0; v < 8; ++v) acc[i][v] = 0.0f;

  for (int kk = 0; kk < DMODEL; kk += 32) {
    // Stage Z tile 128x32 via async global->LDS (512 x 16B chunks, 2/thread)
#pragma unroll
    for (int i = 0; i < 2; ++i) {
      int idx = i * 256 + t;
      int r = idx >> 2, c = (idx & 3) << 3;
      async_copy_b128(Zw + (size_t)(row0 + r) * DMODEL + kk + c, &sA[r][c]);
    }
    // Stage W_O tile 64x32: gather-transpose from [H,D,DH], vectorized along dh
#pragma unroll
    for (int i = 0; i < 2; ++i) {
      int chunk = i * 256 + t;
      int r = chunk >> 3, c = (chunk & 7) << 2;
      int j = kk + c;
      stage4(&WO[(size_t)(j >> 6) * DMODEL * DHEAD + (size_t)(col0 + r) * DHEAD + (j & 63)],
             &sB[r][c]);
    }
    wait_asynccnt0();
    __syncthreads();

    v16bf a = load_frag_a(&sA[wave * 16][0], 40, lane);
#pragma unroll
    for (int nb = 0; nb < 4; ++nb) {
      v16bf b = load_frag_b_nk(&sB[nb * 16][0], 40, lane);
      acc[nb] = wmma_bf16(a, b, acc[nb]);
    }
    __syncthreads();
  }

  const int n = lane & 15, half = lane >> 4;
#pragma unroll
  for (int nb = 0; nb < 4; ++nb) {
    int col  = col0 + nb * 16 + n;
    float bv = bO[col];
#pragma unroll
    for (int v = 0; v < 8; ++v) {
      ef[wave][(half * 8 + v) * 68 + nb * 16 + n] = acc[nb][v] + bv;
    }
  }
  wait_dscnt0();
  {
    int r = lane & 15, c0 = half << 5;   // lane streams 32 contiguous floats
    float* dst = out + (size_t)(row0 + wave * 16 + r) * DMODEL + col0 + c0;
    const float* src = &ef[wave][r * 68 + c0];
#pragma unroll
    for (int i = 0; i < 8; ++i) async_store_b128(dst + 4 * i, src + 4 * i);
  }
  wait_asynccnt0();
}

// ---------------------------------------------------------------------------
// Launcher
// ---------------------------------------------------------------------------
extern "C" void kernel_launch(void* const* d_in, const int* in_sizes, int n_in,
                              void* d_out, int out_size, void* d_ws, size_t ws_size,
                              hipStream_t stream) {
  (void)in_sizes; (void)n_in; (void)out_size; (void)ws_size;
  const float* x  = (const float*)d_in[0];
  // d_in[1] = pos_embed: unused (reference: already in residual stream)
  const float* WQ = (const float*)d_in[2];
  const float* bQ = (const float*)d_in[3];
  const float* WK = (const float*)d_in[4];
  const float* bK = (const float*)d_in[5];
  const float* WV = (const float*)d_in[6];
  const float* bV = (const float*)d_in[7];
  const float* WO = (const float*)d_in[8];
  const float* bO = (const float*)d_in[9];
  float* out = (float*)d_out;

  const size_t mat = (size_t)NTOK * DMODEL;  // 6.29M elems (bf16 -> 12.6 MB)
  __bf16* Qw = (__bf16*)d_ws;
  __bf16* Kw = Qw + mat;
  __bf16* Vt = Kw + mat;   // transposed V: [b][h*64+dh][s]
  __bf16* Zw = Vt + mat;   // total ws use: 4 * 12.6 MB ~= 50 MB

  dim3 g1(NTOK / 128, DMODEL / 64, 3);
  qkv_proj_kernel<<<g1, 256, 0, stream>>>(x, WQ, bQ, WK, bK, WV, bV, Qw, Kw, Vt);

  dim3 g2(SEQ / 64, NHEAD, NBATCH);
  flash_attn_kernel<<<g2, 128, 0, stream>>>(Qw, Kw, Vt, Zw);

  dim3 g3(NTOK / 128, DMODEL / 64, 1);
  out_proj_kernel<<<g3, 256, 0, stream>>>(Zw, WO, bO, out);
}
